// NetGINE_78941498901137
// MI455X (gfx1250) — compile-verified, matrix operand order checked by reading, not run on hardware
//
#include <hip/hip_runtime.h>
#include <hip/hip_bf16.h>

typedef __attribute__((ext_vector_type(16))) _Float16 v16h;
typedef __attribute__((ext_vector_type(8)))  _Float16 v8h;
typedef __attribute__((ext_vector_type(8)))  float    v8f;

#define BN_EPS 1e-5f

__device__ inline v8f zero8() {
    v8f z;
#pragma unroll
    for (int i = 0; i < 8; i++) z[i] = 0.f;
    return z;
}

// Load a 16-half WMMA A/B fragment: halves [k0..k0+7] and [k0+16..k0+23]
__device__ inline v16h load_frag(const _Float16* base, int k0) {
    v8h lo = *(const v8h*)(base + k0);
    v8h hi = *(const v8h*)(base + k0 + 16);
    v16h r;
#pragma unroll
    for (int i = 0; i < 8; i++) { r[i] = lo[i]; r[i + 8] = hi[i]; }
    return r;
}

// ---------------------------------------------------------------------------
// Weight prep kernels
// ---------------------------------------------------------------------------
__global__ void pad_t_f16(const float* __restrict__ src, _Float16* __restrict__ dst,
                          int sk, int sn) {
    int i = blockIdx.x * blockDim.x + threadIdx.x;
    if (i >= 64 * 64) return;
    int n = i >> 6, k = i & 63;
    float v = (k < sk && n < sn) ? src[k * sn + n] : 0.f;
    dst[n * 64 + k] = (_Float16)v;   // transposed, padded
}

__global__ void pad_w1(const float* __restrict__ src, float* __restrict__ dst, int sn) {
    int i = blockIdx.x * blockDim.x + threadIdx.x;
    if (i >= 3 * 64) return;
    int k = i >> 6, n = i & 63;
    dst[i] = (n < sn) ? src[k * sn + n] : 0.f;
}

__global__ void pad_vec64(const float* __restrict__ src, float* __restrict__ dst, int s) {
    int i = threadIdx.x;
    if (i < 64) dst[i] = (i < s) ? src[i] : 0.f;
}

// ---------------------------------------------------------------------------
// Edge kernel: e = relu(ea@W1+b1)@W2+b2 ; msg = relu(x[src]+e)*w ; agg[dst]+=msg
// 64 edges per block, 128 threads (4 waves), WMMA f32_16x16x32_f16.
// FULL=true: no tail guards (exec stays all-ones through the epilogue).
// ---------------------------------------------------------------------------
template <bool FULL>
__global__ __launch_bounds__(128) void edge_msg_kernel(
    const float* __restrict__ ea, const float* __restrict__ ew,
    const long long* __restrict__ srcI, const long long* __restrict__ dstI,
    const float* __restrict__ xin, int xstride,
    const float* __restrict__ w1, const float* __restrict__ b1,
    const _Float16* __restrict__ w2t, const float* __restrict__ b2,
    float* __restrict__ agg, int outd, long long ebase, long long E)
{
    __shared__ __align__(16) _Float16 Ah[64 * 72];
    __shared__ long long Ssrc[64];
    __shared__ long long Sdst[64];
    __shared__ float     Sw[64];

    const int tid = threadIdx.x;
    const long long e0 = ebase + (long long)blockIdx.x * 64;

    if (tid == 0) __builtin_prefetch(ea + (e0 + 64) * 3, 0, 1);

    // Stage edge indices/weights into LDS once per block
    if (tid < 64) {
        long long e = e0 + tid;
        bool ok = FULL || (e < E);
        Ssrc[tid] = ok ? srcI[e] : 0;
        Sdst[tid] = ok ? dstI[e] : 0;
        Sw[tid]   = ok ? ew[e]   : 0.f;
    }

    // Stage 1: hidden = relu(ea @ W1 + b1) -> LDS (f16); thread owns a row-half
    {
        int r  = tid >> 1;
        int c0 = (tid & 1) * 32;
        long long e = e0 + r;
        float a0 = 0.f, a1 = 0.f, a2 = 0.f;
        if (FULL || e < E) { a0 = ea[e * 3]; a1 = ea[e * 3 + 1]; a2 = ea[e * 3 + 2]; }
#pragma unroll 8
        for (int c = c0; c < c0 + 32; c++) {
            float acc = fmaf(a0, w1[c], fmaf(a1, w1[64 + c], fmaf(a2, w1[128 + c], b1[c])));
            Ah[r * 72 + c] = (_Float16)fmaxf(acc, 0.f);
        }
    }
    __syncthreads();

    const int wave = tid >> 5, lane = tid & 31;
    const int half = lane >> 4, mrow = lane & 15;

    // Preload B fragments (constant weights), transposed storage [n][k]
    v16h bf[4][2];
#pragma unroll
    for (int t = 0; t < 4; t++)
#pragma unroll
        for (int ks = 0; ks < 2; ks++)
            bf[t][ks] = load_frag(w2t + (t * 16 + mrow) * 64, ks * 32 + half * 8);

    v8f acc[4];
#pragma unroll
    for (int t = 0; t < 4; t++) acc[t] = zero8();

#pragma unroll
    for (int ks = 0; ks < 2; ks++) {
        v16h a = load_frag(Ah + (wave * 16 + mrow) * 72, ks * 32 + half * 8);
#pragma unroll
        for (int t = 0; t < 4; t++)
            acc[t] = __builtin_amdgcn_wmma_f32_16x16x32_f16(
                false, a, false, bf[t][ks], (short)0, acc[t], false, false);
    }

    // Epilogue: lane covers rows v + half*8, col = t*16 + mrow
    long long sv[8], dv[8];
    float wv[8];
#pragma unroll
    for (int v = 0; v < 8; v++) {
        int r = wave * 16 + v + half * 8;
        sv[v] = Ssrc[r];                 // ds broadcast reads
        dv[v] = Sdst[r];
        wv[v] = Sw[r];
    }
#pragma unroll
    for (int t = 0; t < 4; t++) {
        int col = t * 16 + mrow;
        if (col >= outd) continue;
        float bias = b2[col];
#pragma unroll
        for (int v = 0; v < 8; v++) {
            if (!FULL && (e0 + wave * 16 + v + half * 8) >= E) continue;
            float ev  = acc[t][v] + bias;
            float xv  = xin[sv[v] * (long long)xstride + col];
            float msg = fmaxf(xv + ev, 0.f) * wv[v];
            atomicAdd(&agg[dv[v] * 64 + col], msg);
        }
    }
}

// ---------------------------------------------------------------------------
// Node MLP kernel: h=(1+eps)*x+agg ; t=relu(h@M1+b1)@M2+b2 ; BN stats
// ---------------------------------------------------------------------------
template <bool FULL>
__global__ __launch_bounds__(128) void node_mlp_kernel(
    const float* __restrict__ xin, int xstride, int xcols,
    const float* __restrict__ agg, const float* __restrict__ epsp,
    const _Float16* __restrict__ w1t, const float* __restrict__ b1,
    const _Float16* __restrict__ w2t, const float* __restrict__ b2,
    float* __restrict__ tmp, float* __restrict__ gsum, float* __restrict__ gsq,
    long long nbase, long long nN)
{
    __shared__ __align__(16) _Float16 Ah[64 * 72];
    __shared__ float ssum[64], ssq[64];
    const int tid = threadIdx.x;
    const long long n0 = nbase + (long long)blockIdx.x * 64;
    const float epv = 1.f + epsp[0];

    if (tid < 64) { ssum[tid] = 0.f; ssq[tid] = 0.f; }

    for (int i = tid; i < 64 * 64; i += 128) {
        int r = i >> 6, c = i & 63;
        long long n = n0 + r;
        float hv = 0.f;
        if (FULL || n < nN) {
            float xv = (c < xcols) ? xin[n * (long long)xstride + c] : 0.f;
            hv = epv * xv + agg[n * 64 + c];
        }
        Ah[r * 72 + c] = (_Float16)hv;
    }
    __syncthreads();

    const int wave = tid >> 5, lane = tid & 31;
    const int half = lane >> 4, mrow = lane & 15;

    // GEMM1
    v16h bf[4][2];
#pragma unroll
    for (int t = 0; t < 4; t++)
#pragma unroll
        for (int ks = 0; ks < 2; ks++)
            bf[t][ks] = load_frag(w1t + (t * 16 + mrow) * 64, ks * 32 + half * 8);

    v8f acc[4];
#pragma unroll
    for (int t = 0; t < 4; t++) acc[t] = zero8();
#pragma unroll
    for (int ks = 0; ks < 2; ks++) {
        v16h a = load_frag(Ah + (wave * 16 + mrow) * 72, ks * 32 + half * 8);
#pragma unroll
        for (int t = 0; t < 4; t++)
            acc[t] = __builtin_amdgcn_wmma_f32_16x16x32_f16(
                false, a, false, bf[t][ks], (short)0, acc[t], false, false);
    }
    __syncthreads();

    // relu(acc + b1) back to LDS for GEMM2
#pragma unroll
    for (int t = 0; t < 4; t++) {
        int col = t * 16 + mrow;
        float bias = b1[col];
#pragma unroll
        for (int v = 0; v < 8; v++) {
            int row = wave * 16 + v + half * 8;
            Ah[row * 72 + col] = (_Float16)fmaxf(acc[t][v] + bias, 0.f);
        }
    }
    __syncthreads();

    // GEMM2
#pragma unroll
    for (int t = 0; t < 4; t++)
#pragma unroll
        for (int ks = 0; ks < 2; ks++)
            bf[t][ks] = load_frag(w2t + (t * 16 + mrow) * 64, ks * 32 + half * 8);
#pragma unroll
    for (int t = 0; t < 4; t++) acc[t] = zero8();
#pragma unroll
    for (int ks = 0; ks < 2; ks++) {
        v16h a = load_frag(Ah + (wave * 16 + mrow) * 72, ks * 32 + half * 8);
#pragma unroll
        for (int t = 0; t < 4; t++)
            acc[t] = __builtin_amdgcn_wmma_f32_16x16x32_f16(
                false, a, false, bf[t][ks], (short)0, acc[t], false, false);
    }

#pragma unroll
    for (int t = 0; t < 4; t++) {
        int col = t * 16 + mrow;
        float bias = b2[col];
#pragma unroll
        for (int v = 0; v < 8; v++) {
            int row = wave * 16 + v + half * 8;
            long long n = n0 + row;
            if (FULL || n < nN) {
                float o = acc[t][v] + bias;
                tmp[n * 64 + col] = o;
                atomicAdd(&ssum[col], o);          // ds_add_f32
                atomicAdd(&ssq[col], o * o);
            }
        }
    }
    __syncthreads();
    if (tid < 64) { atomicAdd(&gsum[tid], ssum[tid]); atomicAdd(&gsq[tid], ssq[tid]); }
}

// ---------------------------------------------------------------------------
// BatchNorm + ReLU apply
// ---------------------------------------------------------------------------
__global__ void bn_relu_kernel(const float* __restrict__ in, float* __restrict__ out,
                               long long count, const float* __restrict__ gsum,
                               const float* __restrict__ gsq,
                               const float* __restrict__ g, const float* __restrict__ b,
                               float invR, int ostride)
{
    long long i = (long long)blockIdx.x * blockDim.x + threadIdx.x;
    if (i >= count) return;
    int f = (int)(i & 63);
    long long n = i >> 6;
    float m = gsum[f] * invR;
    float var = fmaxf(gsq[f] * invR - m * m, 0.f);
    float y = (in[i] - m) * rsqrtf(var + BN_EPS) * g[f] + b[f];
    out[n * (long long)ostride + f] = fmaxf(y, 0.f);
}

// ---------------------------------------------------------------------------
// Readout helpers
// ---------------------------------------------------------------------------
__global__ void seg_sum_kernel(const float* __restrict__ in, const long long* __restrict__ idx,
                               float* __restrict__ out, long long total, int cshift)
{
    long long i = (long long)blockIdx.x * blockDim.x + threadIdx.x;
    if (i >= total) return;
    long long n = i >> cshift;
    int f = (int)(i & ((1 << cshift) - 1));
    atomicAdd(&out[(idx[n] << cshift) + f], in[i]);
}

__global__ void count_seg_kernel(const long long* __restrict__ idx, float* __restrict__ cnt, int n)
{
    int i = blockIdx.x * blockDim.x + threadIdx.x;
    if (i < n) atomicAdd(&cnt[idx[i]], 1.f);
}

__global__ void div_rows_kernel(float* __restrict__ buf, const float* __restrict__ cnt,
                                long long total, int cshift)
{
    long long i = (long long)blockIdx.x * blockDim.x + threadIdx.x;
    if (i >= total) return;
    buf[i] /= fmaxf(cnt[i >> cshift], 1.f);
}

__global__ void fc_bn_pre_kernel(const float* __restrict__ in, const float* __restrict__ w,
                                 const float* __restrict__ bias, float* __restrict__ out,
                                 float* __restrict__ gsum, float* __restrict__ gsq,
                                 int rows, int K)
{
    int i = blockIdx.x * blockDim.x + threadIdx.x;
    if (i >= rows * 64) return;
    int r = i >> 6, f = i & 63;
    float s = bias[f];
    for (int k = 0; k < K; k++) s = fmaf(in[r * K + k], w[k * 64 + f], s);
    out[i] = s;
    atomicAdd(&gsum[f], s);
    atomicAdd(&gsq[f], s * s);
}

__global__ void final_fc3_kernel(const float* __restrict__ sbuf, const float* __restrict__ scnt,
                                 const float* __restrict__ w, const float* __restrict__ b,
                                 float* __restrict__ out, int S)
{
    int s = blockIdx.x * blockDim.x + threadIdx.x;
    if (s >= S) return;
    float inv = 1.f / fmaxf(scnt[s], 1.f);
    float acc = b[0];
    for (int f = 0; f < 64; f++) acc = fmaf(sbuf[s * 64 + f] * inv, w[f], acc);
    out[s] = acc;
}

// ---------------------------------------------------------------------------
// Host launcher
// ---------------------------------------------------------------------------
extern "C" void kernel_launch(void* const* d_in, const int* in_sizes, int n_in,
                              void* d_out, int out_size, void* d_ws, size_t ws_size,
                              hipStream_t stream)
{
    const float* x        = (const float*)d_in[0];
    const float* ea       = (const float*)d_in[1];
    const float* ew       = (const float*)d_in[2];
    const long long* ei   = (const long long*)d_in[3];
    const long long* bat  = (const long long*)d_in[4];
    const long long* igi  = (const long long*)d_in[5];

    const long long N = in_sizes[0] / 28;
    const long long E = in_sizes[2];
    const int G = in_sizes[5];
    const int S = out_size;

    // ---- workspace carve-up ----
    size_t off = 0;
    auto alloc = [&](size_t bytes) -> char* {
        off = (off + 255) & ~(size_t)255;
        char* p = (char*)d_ws + off;
        off += bytes;
        return p;
    };
    float* h_cat  = (float*)alloc((size_t)N * 256 * 4);
    float* agg    = (float*)alloc((size_t)N * 64 * 4);
    float* tmp    = (float*)alloc((size_t)N * 64 * 4);
    float* gsum   = (float*)alloc(64 * 4);
    float* gsq    = (float*)alloc(64 * 4);
    float* pooled = (float*)alloc((size_t)G * 256 * 4);
    float* gcnt   = (float*)alloc((size_t)G * 4);
    float* gb1p   = (float*)alloc((size_t)G * 64 * 4);
    float* gb1    = (float*)alloc((size_t)G * 64 * 4);
    float* gb2p   = (float*)alloc((size_t)G * 64 * 4);
    float* gb2    = (float*)alloc((size_t)G * 64 * 4);
    float* sbuf   = (float*)alloc((size_t)S * 64 * 4);
    float* scnt   = (float*)alloc((size_t)S * 4);

    struct LW { float *ew1, *eb1, *eb2, *mb1, *mb2; _Float16 *ew2t, *mw1t, *mw2t; } lw[4];
    for (int l = 0; l < 4; l++) {
        lw[l].ew1  = (float*)alloc(3 * 64 * 4);
        lw[l].eb1  = (float*)alloc(64 * 4);
        lw[l].eb2  = (float*)alloc(64 * 4);
        lw[l].mb1  = (float*)alloc(64 * 4);
        lw[l].mb2  = (float*)alloc(64 * 4);
        lw[l].ew2t = (_Float16*)alloc(64 * 64 * 2);
        lw[l].mw1t = (_Float16*)alloc(64 * 64 * 2);
        lw[l].mw2t = (_Float16*)alloc(64 * 64 * 2);
    }

    // ---- weight prep ----
    for (int l = 0; l < 4; l++) {
        const float* bw1 = l == 0 ? (const float*)d_in[6]  : (const float*)d_in[17] + (l - 1) * 3 * 64;
        const float* bb1 = l == 0 ? (const float*)d_in[7]  : (const float*)d_in[18] + (l - 1) * 64;
        const float* bw2 = l == 0 ? (const float*)d_in[8]  : (const float*)d_in[19] + (l - 1) * 4096;
        const float* bb2 = l == 0 ? (const float*)d_in[9]  : (const float*)d_in[20] + (l - 1) * 64;
        const float* mw1 = l == 0 ? (const float*)d_in[10] : (const float*)d_in[21] + (l - 1) * 4096;
        const float* mb1 = l == 0 ? (const float*)d_in[11] : (const float*)d_in[22] + (l - 1) * 64;
        const float* mw2 = l == 0 ? (const float*)d_in[12] : (const float*)d_in[23] + (l - 1) * 4096;
        const float* mb2 = l == 0 ? (const float*)d_in[13] : (const float*)d_in[24] + (l - 1) * 64;
        int d = (l == 0) ? 28 : 64;

        pad_w1<<<1, 192, 0, stream>>>(bw1, lw[l].ew1, d);
        pad_vec64<<<1, 64, 0, stream>>>(bb1, lw[l].eb1, d);
        pad_vec64<<<1, 64, 0, stream>>>(bb2, lw[l].eb2, d);
        pad_vec64<<<1, 64, 0, stream>>>(mb1, lw[l].mb1, d);
        pad_vec64<<<1, 64, 0, stream>>>(mb2, lw[l].mb2, 64);
        pad_t_f16<<<16, 256, 0, stream>>>(bw2, lw[l].ew2t, d, d);
        pad_t_f16<<<16, 256, 0, stream>>>(mw1, lw[l].mw1t, d, d);
        pad_t_f16<<<16, 256, 0, stream>>>(mw2, lw[l].mw2t, d, 64);
    }

    const long long EBfull = E / 64;   // full 64-edge tiles
    const long long Etail  = E - EBfull * 64;
    const long long NBfull = N / 64;
    const long long Ntail  = N - NBfull * 64;

    // ---- 4 GINE blocks ----
    for (int l = 0; l < 4; l++) {
        const float* epsp = l == 0 ? (const float*)d_in[14] : (const float*)d_in[25] + (l - 1);
        const float* bng  = l == 0 ? (const float*)d_in[15] : (const float*)d_in[26] + (l - 1) * 64;
        const float* bnb  = l == 0 ? (const float*)d_in[16] : (const float*)d_in[27] + (l - 1) * 64;
        const float* xin  = l == 0 ? x : h_cat + (l - 1) * 64;
        const int xstride = l == 0 ? 28 : 256;
        const int outd    = l == 0 ? 28 : 64;

        hipMemsetAsync(agg, 0, (size_t)N * 64 * 4, stream);
        hipMemsetAsync(gsum, 0, 64 * 4, stream);
        hipMemsetAsync(gsq, 0, 64 * 4, stream);

        if (EBfull > 0)
            edge_msg_kernel<true><<<dim3((unsigned)EBfull), dim3(128), 0, stream>>>(
                ea, ew, ei, ei + E, xin, xstride,
                lw[l].ew1, lw[l].eb1, lw[l].ew2t, lw[l].eb2, agg, outd, 0, E);
        if (Etail > 0)
            edge_msg_kernel<false><<<dim3(1), dim3(128), 0, stream>>>(
                ea, ew, ei, ei + E, xin, xstride,
                lw[l].ew1, lw[l].eb1, lw[l].ew2t, lw[l].eb2, agg, outd, EBfull * 64, E);

        if (NBfull > 0)
            node_mlp_kernel<true><<<dim3((unsigned)NBfull), dim3(128), 0, stream>>>(
                xin, xstride, outd, agg, epsp,
                lw[l].mw1t, lw[l].mb1, lw[l].mw2t, lw[l].mb2,
                tmp, gsum, gsq, 0, N);
        if (Ntail > 0)
            node_mlp_kernel<false><<<dim3(1), dim3(128), 0, stream>>>(
                xin, xstride, outd, agg, epsp,
                lw[l].mw1t, lw[l].mb1, lw[l].mw2t, lw[l].mb2,
                tmp, gsum, gsq, NBfull * 64, N);

        bn_relu_kernel<<<dim3((unsigned)((N * 64 + 255) / 256)), dim3(256), 0, stream>>>(
            tmp, h_cat + l * 64, N * 64, gsum, gsq, bng, bnb, 1.f / (float)N, 256);
    }

    // ---- graph pooling + FC head ----
    hipMemsetAsync(pooled, 0, (size_t)G * 256 * 4, stream);
    hipMemsetAsync(gcnt, 0, (size_t)G * 4, stream);
    seg_sum_kernel<<<dim3((unsigned)((N * 256 + 255) / 256)), dim3(256), 0, stream>>>(
        h_cat, bat, pooled, N * 256, 8);
    count_seg_kernel<<<dim3((unsigned)((N + 255) / 256)), dim3(256), 0, stream>>>(bat, gcnt, (int)N);
    div_rows_kernel<<<dim3((unsigned)(((long long)G * 256 + 255) / 256)), dim3(256), 0, stream>>>(
        pooled, gcnt, (long long)G * 256, 8);

    hipMemsetAsync(gsum, 0, 64 * 4, stream);
    hipMemsetAsync(gsq, 0, 64 * 4, stream);
    fc_bn_pre_kernel<<<dim3((G * 64 + 255) / 256), dim3(256), 0, stream>>>(
        pooled, (const float*)d_in[28], (const float*)d_in[29], gb1p, gsum, gsq, G, 256);
    bn_relu_kernel<<<dim3((unsigned)(((long long)G * 64 + 255) / 256)), dim3(256), 0, stream>>>(
        gb1p, gb1, (long long)G * 64, gsum, gsq,
        (const float*)d_in[30], (const float*)d_in[31], 1.f / (float)G, 64);

    hipMemsetAsync(gsum, 0, 64 * 4, stream);
    hipMemsetAsync(gsq, 0, 64 * 4, stream);
    fc_bn_pre_kernel<<<dim3((G * 64 + 255) / 256), dim3(256), 0, stream>>>(
        gb1, (const float*)d_in[32], (const float*)d_in[33], gb2p, gsum, gsq, G, 64);
    bn_relu_kernel<<<dim3((unsigned)(((long long)G * 64 + 255) / 256)), dim3(256), 0, stream>>>(
        gb2p, gb2, (long long)G * 64, gsum, gsq,
        (const float*)d_in[34], (const float*)d_in[35], 1.f / (float)G, 64);

    hipMemsetAsync(sbuf, 0, (size_t)S * 64 * 4, stream);
    hipMemsetAsync(scnt, 0, (size_t)S * 4, stream);
    seg_sum_kernel<<<dim3((unsigned)(((long long)G * 64 + 255) / 256)), dim3(256), 0, stream>>>(
        gb2, igi, sbuf, (long long)G * 64, 6);
    count_seg_kernel<<<dim3((G + 255) / 256), dim3(256), 0, stream>>>(igi, scnt, G);

    final_fc3_kernel<<<dim3((S + 63) / 64), dim3(64), 0, stream>>>(
        sbuf, scnt, (const float*)d_in[36], (const float*)d_in[37], (float*)d_out, S);
}
